// TriangularSelfAttentionBlock_71571335020725
// MI455X (gfx1250) — matrix operand order repaired and checked by reading, not compiled, not run on hardware
//
#include <hip/hip_runtime.h>
#include <hip/hip_bf16.h>

// ---------------------------------------------------------------------------
// TriangularSelfAttentionBlock for MI455X (gfx1250, wave32, WMMA f16 16x16x32)
// ---------------------------------------------------------------------------

typedef _Float16 h16;
typedef __attribute__((ext_vector_type(16))) _Float16 v16h;
typedef __attribute__((ext_vector_type(8)))  _Float16 v8h;
typedef __attribute__((ext_vector_type(8)))  float    v8f;

#define L_   256
#define S_   1024
#define P_   128
#define LL_  65536

__device__ __forceinline__ v8f wmma32(v16h a, v16h b, v8f c) {
  // v_wmma_f32_16x16x32_f16 : D(16x16 f32) = A(16x32 f16) x B(32x16 f16) + C
  return __builtin_amdgcn_wmma_f32_16x16x32_f16(false, a, false, b, (short)0, c,
                                                false, false);
}

__device__ __forceinline__ v8f zero8() {
  v8f r;
#pragma unroll
  for (int i = 0; i < 8; ++i) r[i] = 0.f;
  return r;
}

// A fragment (16x32 f16, row-major source, lda in elements).
// lane 0-15: row m0+lane, K = {0..7, 16..23}; lane 16-31: same rows, K = {8..15, 24..31}
__device__ __forceinline__ v16h ld_a(const h16* A, int lda, int m0, int k0, int lane) {
  const h16* p = A + (size_t)(m0 + (lane & 15)) * (size_t)lda +
                 (size_t)(k0 + ((lane >> 4) << 3));
  v8h lo = *(const v8h*)p;
  v8h hi = *(const v8h*)(p + 16);
  v16h r;
#pragma unroll
  for (int i = 0; i < 8; ++i) { r[i] = lo[i]; r[i + 8] = hi[i]; }
  return r;
}

// B fragment (32x16 f16) from K-major storage Bt[k][n] (ldb elements per k row).
// lane 0-15: col n0+lane, K = {0..7, 16..23}; lane 16-31: same cols, K = {8..15, 24..31}
__device__ __forceinline__ v16h ld_b(const h16* Bt, int ldb, int n0, int k0, int lane) {
  const h16* p = Bt + (size_t)(k0 + ((lane >> 4) << 3)) * (size_t)ldb + (size_t)(n0 + (lane & 15));
  v16h r;
#pragma unroll
  for (int i = 0; i < 8; ++i) { r[i] = p[(size_t)i * ldb]; r[i + 8] = p[(size_t)(16 + i) * ldb]; }
  return r;
}

// ---------------------------------------------------------------------------
// Generic GEMM: C[M,N] = act(A[M,K](f16) * Bt[K,N](f16) + bias)
// 4 waves / block, each wave a 16x16 tile. ACT: 0 none, 1 relu, 2 sigmoid.
// ---------------------------------------------------------------------------
template <int ACT, bool HOUT, bool TOUT, bool ADD>
__global__ __launch_bounds__(128)
void k_gemm(const h16* __restrict__ A, int lda, const h16* __restrict__ Bt, int ldb,
            const float* __restrict__ bias, void* __restrict__ Cv, int ldc,
            int M, int N, int K) {
  int lane = threadIdx.x & 31, wv = threadIdx.x >> 5;
  int m0 = blockIdx.x * 16;
  int n0 = (blockIdx.y * 4 + wv) * 16;
  if (n0 >= N) return;
  v8f acc = zero8();
  for (int k0 = 0; k0 < K; k0 += 32) {
    v16h a = ld_a(A, lda, m0, k0, lane);
    v16h b = ld_b(Bt, ldb, n0, k0, lane);
    if (k0 + 32 < K)  // gfx1250 global_prefetch_b8 of next A chunk
      __builtin_prefetch(A + (size_t)(m0 + (lane & 15)) * (size_t)lda + k0 + 32, 0, 1);
    acc = wmma32(a, b, acc);
  }
  int nn = n0 + (lane & 15);
  float bv = (bias && nn < N) ? bias[nn] : 0.f;
#pragma unroll
  for (int v = 0; v < 8; ++v) {
    int mm = m0 + v + ((lane >> 4) << 3);
    if (mm < M && nn < N) {
      float x = acc[v] + bv;
      if (ACT == 1) x = fmaxf(x, 0.f);
      if (ACT == 2) x = 1.f / (1.f + __expf(-x));
      size_t idx = TOUT ? (size_t)nn * (size_t)ldc + (size_t)mm
                        : (size_t)mm * (size_t)ldc + (size_t)nn;
      if (HOUT) {
        ((h16*)Cv)[idx] = (h16)x;
      } else {
        float* Cf = (float*)Cv;
        if (ADD) Cf[idx] += x; else Cf[idx] = x;
      }
    }
  }
}

// ---------------------------------------------------------------------------
// LayerNorm over last dim D (f32 in -> f16 out), one block (128 thr) per row
// ---------------------------------------------------------------------------
__global__ __launch_bounds__(128)
void k_ln(const float* __restrict__ x, const float* __restrict__ g,
          const float* __restrict__ b, h16* __restrict__ y, int D) {
  __shared__ float sh[2][128];
  int row = blockIdx.x, t = threadIdx.x;
  const float* xr = x + (size_t)row * D;
  float s = 0.f, ss = 0.f;
  for (int j = t; j < D; j += 128) { float v = xr[j]; s += v; ss += v * v; }
  sh[0][t] = s; sh[1][t] = ss; __syncthreads();
  for (int k = 64; k > 0; k >>= 1) {
    if (t < k) { sh[0][t] += sh[0][t + k]; sh[1][t] += sh[1][t + k]; }
    __syncthreads();
  }
  float mean = sh[0][0] / D;
  float var  = sh[1][0] / D - mean * mean;
  float inv  = rsqrtf(var + 1e-5f);
  h16* yr = y + (size_t)row * D;
  for (int j = t; j < D; j += 128) yr[j] = (h16)((xr[j] - mean) * inv * g[j] + b[j]);
}

// ---------------------------------------------------------------------------
// Elementwise / layout kernels
// ---------------------------------------------------------------------------
#define GS(i, n) for (size_t i = (size_t)blockIdx.x * blockDim.x + threadIdx.x; \
                      i < (n); i += (size_t)gridDim.x * blockDim.x)

// f32 weight [N][K] -> f16 K-major [K][ldn] (zero-padded n >= N)
__global__ void k_cvt_w(const float* __restrict__ w, h16* __restrict__ wt,
                        int N, int K, int ldn) {
  size_t tot = (size_t)K * (size_t)ldn;
  GS(i, tot) {
    int k = (int)(i / (size_t)ldn), n = (int)(i % (size_t)ldn);
    wt[i] = (n < N) ? (h16)w[(size_t)n * K + k] : (h16)0.f;
  }
}
__global__ void k_mul16(const h16* a, const h16* b, h16* o, size_t n) {
  GS(i, n) o[i] = (h16)((float)a[i] * (float)b[i]);
}
__global__ void k_fma32(float* z, const float* u, const h16* g, size_t n) {
  GS(i, n) z[i] += u[i] * (float)g[i];
}
// dst[(r*L+s)*P+c] = src[(s*L+r)*P+c]   (pair transpose)
__global__ void k_swap_copy(const float* src, float* dst) {
  GS(i, (size_t)LL_ * P_) {
    int r = (int)(i >> 15), s2 = ((int)i >> 7) & 255, c = (int)i & 127;
    dst[i] = src[((size_t)s2 * 256 + r) * 128 + c];
  }
}
__global__ void k_swap_add(float* z, const float* u) {
  GS(i, (size_t)LL_ * P_) {
    int r = (int)(i >> 15), s2 = ((int)i >> 7) & 255, c = (int)i & 127;
    z[i] += u[((size_t)s2 * 256 + r) * 128 + c];
  }
}
// out[c][r][s] = mask * sigmoid_gate * proj at token t (t = r*L+s or s*L+r)
__global__ void k_combine(const h16* g, const h16* p, const float* mask,
                          h16* o, int swap) {
  GS(i, (size_t)P_ * LL_) {
    int c = (int)(i >> 16); int rs = (int)i & 65535;
    int r = rs >> 8, s2 = rs & 255;
    int t = swap ? (s2 << 8) + r : rs;
    float m = mask[t >> 8] * mask[t & 255];
    o[i] = (h16)(m * (float)g[(size_t)t * 128 + c] * (float)p[(size_t)t * 128 + c]);
  }
}
// SequenceToPair features: prod = qh[j]*kh[i], diff = qh[j]-kh[i]
__global__ void k_s2p_feat(const float* sp, h16* F) {
  GS(i, (size_t)LL_ * P_) {
    int t = (int)(i >> 7), c = (int)i & 127;
    int ii = t >> 8, jj = t & 255;
    float v = (c < 64) ? sp[jj * 128 + c] * sp[ii * 128 + 64 + c]
                       : sp[jj * 128 + (c - 64)] - sp[ii * 128 + 64 + (c - 64)];
    F[i] = (h16)v;
  }
}
// seq attention K transpose: kts[(h*64+d)*256+t] = qkv[t][h*192+64+d]
__global__ void k_seq_kt(const h16* qkv, h16* kts) {
  GS(i, (size_t)16 * 64 * 256) {
    int h = (int)(i >> 14), d = ((int)i >> 8) & 63, t = (int)i & 255;
    kts[i] = qkv[(size_t)t * 3072 + h * 192 + 64 + d];
  }
}

// ---------------------------------------------------------------------------
// Triangle multiplicative einsum: x[i,j,c] = sum_k a[c][i][k] * bT[c][k][j]
// (channel-major f16 operands, 128 independent 256x256x256 WMMA GEMMs)
// ---------------------------------------------------------------------------
__global__ __launch_bounds__(128)
void k_trimul(const h16* __restrict__ a, const h16* __restrict__ bT,
              float* __restrict__ x) {
  int lane = threadIdx.x & 31, wv = threadIdx.x >> 5;
  int i0 = blockIdx.x * 16, j0 = (blockIdx.y * 4 + wv) * 16, c = blockIdx.z;
  const h16* A = a  + (size_t)c * LL_;
  const h16* B = bT + (size_t)c * LL_;
  v8f acc = zero8();
  for (int k0 = 0; k0 < 256; k0 += 32)
    acc = wmma32(ld_a(A, 256, i0, k0, lane), ld_b(B, 256, j0, k0, lane), acc);
#pragma unroll
  for (int v = 0; v < 8; ++v) {
    int mm = i0 + v + ((lane >> 4) << 3), nn = j0 + (lane & 15);
    x[((size_t)mm * 256 + nn) * 128 + c] = acc[v];
  }
}

// ---------------------------------------------------------------------------
// Triangle attention (flash, online softmax). One wave = (row i, head h, 16 q).
// q [LL][128], kT [128][LL] (chan-major), v [LL][128], bias4 [LL][4]
// ---------------------------------------------------------------------------
__global__ __launch_bounds__(128)
void k_tri_flash(const h16* __restrict__ q, const h16* __restrict__ kT,
                 const h16* __restrict__ v, const float* __restrict__ bias4,
                 const float* __restrict__ mask, h16* __restrict__ outp) {
  __shared__ __align__(16) h16 pt[4][16][32];
  int lane = threadIdx.x & 31, w = threadIdx.x >> 5;
  int i = blockIdx.z, h = blockIdx.y;
  int q0 = (blockIdx.x * 4 + w) * 16;
  const h16* Qb  = q  + (size_t)i * 256 * 128 + h * 32;
  const h16* KTb = kT + (size_t)(h * 32) * LL_ + (size_t)i * 256;
  const h16* Vb  = v  + (size_t)i * 256 * 128 + h * 32;
  v16h aq = ld_a(Qb, 128, q0, 0, lane);
  v8f o0 = zero8(), o1 = zero8();
  float mrow[8], lrow[8], corr[8];
#pragma unroll
  for (int t = 0; t < 8; ++t) { mrow[t] = -1e30f; lrow[t] = 0.f; }
  float mi = mask[i];
  const float scale = 0.17677669529663687f;  // 32^-0.5
  for (int kc = 0; kc < 256; kc += 32) {
    v8f s0 = wmma32(aq, ld_b(KTb, LL_, kc,      0, lane), zero8());
    v8f s1 = wmma32(aq, ld_b(KTb, LL_, kc + 16, 0, lane), zero8());
#pragma unroll
    for (int t = 0; t < 8; ++t) {
      int m  = t + ((lane >> 4) << 3);
      int qq = q0 + m;
      int k0n = kc + (lane & 15), k1n = k0n + 16;
      float x0 = s0[t] * scale + bias4[((size_t)qq * 256 + k0n) * 4 + h] +
                 1e9f * (mi * mask[k0n] - 1.f);
      float x1 = s1[t] * scale + bias4[((size_t)qq * 256 + k1n) * 4 + h] +
                 1e9f * (mi * mask[k1n] - 1.f);
      float mx = fmaxf(x0, x1);
#pragma unroll
      for (int d = 1; d < 16; d <<= 1) mx = fmaxf(mx, __shfl_xor(mx, d, 32));
      float nm = fmaxf(mrow[t], mx);
      float cc = __expf(mrow[t] - nm);
      float p0 = __expf(x0 - nm), p1 = __expf(x1 - nm);
      float rs = p0 + p1;
#pragma unroll
      for (int d = 1; d < 16; d <<= 1) rs += __shfl_xor(rs, d, 32);
      lrow[t] = lrow[t] * cc + rs;
      mrow[t] = nm; corr[t] = cc;
      pt[w][m][lane & 15]        = (h16)p0;
      pt[w][m][(lane & 15) + 16] = (h16)p1;
    }
#pragma unroll
    for (int t = 0; t < 8; ++t) { o0[t] *= corr[t]; o1[t] *= corr[t]; }
    __syncthreads();
    v16h ap = ld_a(&pt[w][0][0], 32, 0, 0, lane);
    o0 = wmma32(ap, ld_b(Vb, 128, 0,  kc, lane), o0);
    o1 = wmma32(ap, ld_b(Vb, 128, 16, kc, lane), o1);
    __syncthreads();
  }
#pragma unroll
  for (int t = 0; t < 8; ++t) {
    int m = t + ((lane >> 4) << 3);
    float inv = 1.f / lrow[t];
    size_t o = ((size_t)i * 256 + q0 + m) * 128 + h * 32;
    outp[o + (lane & 15)]      = (h16)(o0[t] * inv);
    outp[o + 16 + (lane & 15)] = (h16)(o1[t] * inv);
  }
}

// ---------------------------------------------------------------------------
// Sequence attention (flash). 16 heads, head dim 64.
// qkv [256][3072] (per head: q|k|v of 64), kts [1024][256], biasS [LL][16]
// ---------------------------------------------------------------------------
__global__ __launch_bounds__(128)
void k_seq_flash(const h16* __restrict__ qkv, const h16* __restrict__ kts,
                 const float* __restrict__ biasS, const float* __restrict__ mask,
                 h16* __restrict__ outp) {
  __shared__ __align__(16) h16 pt[4][16][32];
  int lane = threadIdx.x & 31, w = threadIdx.x >> 5;
  int h = blockIdx.y;
  int q0 = (blockIdx.x * 4 + w) * 16;
  const h16* Qb = qkv + h * 192;
  const h16* Kb = kts + (size_t)h * 64 * 256;
  const h16* Vb = qkv + h * 192 + 128;
  v16h aq0 = ld_a(Qb, 3072, q0, 0, lane);
  v16h aq1 = ld_a(Qb, 3072, q0, 32, lane);
  v8f o0 = zero8(), o1 = zero8(), o2 = zero8(), o3 = zero8();
  float mrow[8], lrow[8], corr[8];
#pragma unroll
  for (int t = 0; t < 8; ++t) { mrow[t] = -1e30f; lrow[t] = 0.f; }
  const float scale = 0.125f;  // 64^-0.5
  for (int kc = 0; kc < 256; kc += 32) {
    v8f s0 = wmma32(aq0, ld_b(Kb, 256, kc, 0, lane), zero8());
    s0     = wmma32(aq1, ld_b(Kb, 256, kc, 32, lane), s0);
    v8f s1 = wmma32(aq0, ld_b(Kb, 256, kc + 16, 0, lane), zero8());
    s1     = wmma32(aq1, ld_b(Kb, 256, kc + 16, 32, lane), s1);
#pragma unroll
    for (int t = 0; t < 8; ++t) {
      int m  = t + ((lane >> 4) << 3);
      int qq = q0 + m;
      int k0n = kc + (lane & 15), k1n = k0n + 16;
      float x0 = s0[t] * scale + biasS[((size_t)qq * 256 + k0n) * 16 + h];
      float x1 = s1[t] * scale + biasS[((size_t)qq * 256 + k1n) * 16 + h];
      x0 = (mask[k0n] > 0.f) ? x0 : -1e30f;
      x1 = (mask[k1n] > 0.f) ? x1 : -1e30f;
      float mx = fmaxf(x0, x1);
#pragma unroll
      for (int d = 1; d < 16; d <<= 1) mx = fmaxf(mx, __shfl_xor(mx, d, 32));
      float nm = fmaxf(mrow[t], mx);
      float cc = __expf(mrow[t] - nm);
      float p0 = __expf(x0 - nm), p1 = __expf(x1 - nm);
      float rs = p0 + p1;
#pragma unroll
      for (int d = 1; d < 16; d <<= 1) rs += __shfl_xor(rs, d, 32);
      lrow[t] = lrow[t] * cc + rs;
      mrow[t] = nm; corr[t] = cc;
      pt[w][m][lane & 15]        = (h16)p0;
      pt[w][m][(lane & 15) + 16] = (h16)p1;
    }
#pragma unroll
    for (int t = 0; t < 8; ++t) {
      o0[t] *= corr[t]; o1[t] *= corr[t]; o2[t] *= corr[t]; o3[t] *= corr[t];
    }
    __syncthreads();
    v16h ap = ld_a(&pt[w][0][0], 32, 0, 0, lane);
    o0 = wmma32(ap, ld_b(Vb, 3072, 0,  kc, lane), o0);
    o1 = wmma32(ap, ld_b(Vb, 3072, 16, kc, lane), o1);
    o2 = wmma32(ap, ld_b(Vb, 3072, 32, kc, lane), o2);
    o3 = wmma32(ap, ld_b(Vb, 3072, 48, kc, lane), o3);
    __syncthreads();
  }
#pragma unroll
  for (int t = 0; t < 8; ++t) {
    int m = t + ((lane >> 4) << 3);
    float inv = 1.f / lrow[t];
    size_t o = (size_t)(q0 + m) * 1024 + h * 64;
    outp[o + (lane & 15)]      = (h16)(o0[t] * inv);
    outp[o + 16 + (lane & 15)] = (h16)(o1[t] * inv);
    outp[o + 32 + (lane & 15)] = (h16)(o2[t] * inv);
    outp[o + 48 + (lane & 15)] = (h16)(o3[t] * inv);
  }
}

// ---------------------------------------------------------------------------
// Host-side GEMM dispatch
// ---------------------------------------------------------------------------
enum { M_F16 = 0, M_F16T = 1, M_F32 = 2, M_F32ADD = 3 };

static void gemm(hipStream_t st, const h16* A, int lda, const h16* Bt, int ldb,
                 const float* bias, void* C, int ldc, int M, int N, int K,
                 int act, int mode) {
  dim3 g((unsigned)(M / 16), (unsigned)((N + 63) / 64)), b(128);
  switch (mode) {
    case M_F16:
      if (act == 0)      k_gemm<0, true, false, false><<<g, b, 0, st>>>(A, lda, Bt, ldb, bias, C, ldc, M, N, K);
      else if (act == 1) k_gemm<1, true, false, false><<<g, b, 0, st>>>(A, lda, Bt, ldb, bias, C, ldc, M, N, K);
      else               k_gemm<2, true, false, false><<<g, b, 0, st>>>(A, lda, Bt, ldb, bias, C, ldc, M, N, K);
      break;
    case M_F16T:   k_gemm<0, true,  true,  false><<<g, b, 0, st>>>(A, lda, Bt, ldb, bias, C, ldc, M, N, K); break;
    case M_F32:    k_gemm<0, false, false, false><<<g, b, 0, st>>>(A, lda, Bt, ldb, bias, C, ldc, M, N, K); break;
    case M_F32ADD: k_gemm<0, false, false, true ><<<g, b, 0, st>>>(A, lda, Bt, ldb, bias, C, ldc, M, N, K); break;
  }
}

// ---------------------------------------------------------------------------
// kernel_launch: orchestrates the whole block. Param indices follow jax
// pytree flattening (dict keys sorted alphabetically at every level).
// ---------------------------------------------------------------------------
extern "C" void kernel_launch(void* const* d_in, const int* in_sizes, int n_in,
                              void* d_out, int out_size, void* d_ws, size_t ws_size,
                              hipStream_t stream) {
  (void)in_sizes; (void)out_size; (void)ws_size;
  const float* s_in = (const float*)d_in[0];
  const float* z_in = (const float*)d_in[1];
  const float* mask = (const float*)d_in[2];
  const float* PRM[96];
  for (int i = 0; i < n_in && i < 96; ++i) PRM[i] = (const float*)d_in[i];

  // param leaf indices (sorted-key pytree order)
  enum {
    P_ATTN_G_B = 3, P_ATTN_G_W, P_ATTN_O_B, P_ATTN_O_W, P_ATTN_PROJ_W,
    P_LN1_B = 8, P_LN1_G,
    P_MP_LN_B = 10, P_MP_LN_G, P_MP_W1_B, P_MP_W1_W, P_MP_W2_B, P_MP_W2_W,
    P_MS_LN_B = 16, P_MS_LN_G, P_MS_W1_B, P_MS_W1_W, P_MS_W2_B, P_MS_W2_W,
    P_P2S_LIN_W = 22, P_P2S_LN_B, P_P2S_LN_G,
    P_S2P_LN_B = 25, P_S2P_LN_G, P_S2P_O_B, P_S2P_O_W, P_S2P_PROJ_B, P_S2P_PROJ_W,
    T_TAE = 31,  // b.w, g.b, g.w, k.w, ln.b, ln.g, o.b, o.w, q.w, v.w
    T_TAS = 41,
    T_TMI = 51,  // a_g.b, a_g.w, a_p.b, a_p.w, b_g.b, b_g.w, b_p.b, b_p.w,
                 // g.b, g.w, ln_in.b, ln_in.g, ln_out.b, ln_out.g, z.b, z.w
    T_TMO = 67
  };

  float* s_out = (float*)d_out;                    // [256][1024]
  float* z_out = s_out + (size_t)L_ * S_;          // [256][256][128]
  char* ws = (char*)d_ws;
  constexpr size_t SZ = (size_t)LL_ * P_ * 2;      // 16 MB slot
  const size_t OFF_ZN = 0, OFF_T1 = SZ, OFF_T2 = 2 * SZ, OFF_ACM = 3 * SZ,
               OFF_BCM = 4 * SZ, OFF_XT = 5 * SZ /*f32, slots 5-6*/,
               OFF_XN = 7 * SZ, OFF_ZU = 8 * SZ /*f32, slots 8-9*/,
               OFF_G16 = 10 * SZ, OFF_W = 11 * SZ /*slots 11-12*/,
               OFF_MISC = 13 * SZ;

  float* bias4 = (float*)(ws + OFF_MISC);                   // [LL][4]
  float* biasS = (float*)(ws + OFF_MISC + 1048576);         // [LL][16]
  h16*   y16   = (h16*)(ws + OFF_MISC + 5242880);           // [256][1024]
  h16*   qkv   = (h16*)(ws + OFF_MISC + 5767168);           // [256][3072]
  h16*   kts   = (h16*)(ws + OFF_MISC + 7340032);           // [1024][256]
  h16*   oS    = (h16*)(ws + OFF_MISC + 7864320);           // [256][1024]
  h16*   gS    = (h16*)(ws + OFF_MISC + 8388608);           // [256][1024]
  h16*   ogS   = (h16*)(ws + OFF_MISC + 8912896);           // [256][1024]
  h16*   x16s  = (h16*)(ws + OFF_MISC + 9437184);           // [256][1024]
  h16*   h4k   = (h16*)(ws + OFF_MISC + 9961472);           // [256][4096]
  float* sp    = (float*)(ws + OFF_MISC + 12058624);        // [256][128]

  size_t wo = OFF_W;
  auto conv = [&](int idx, int N, int K, int ldn) -> const h16* {
    h16* dst = (h16*)(ws + wo);
    size_t cnt = (size_t)K * ldn;
    wo += (cnt * 2 + 255) & ~(size_t)255;
    k_cvt_w<<<dim3(2048), dim3(256), 0, stream>>>(PRM[idx], dst, N, K, ldn);
    return dst;
  };

  // ---- init outputs with residual bases ----
  hipMemcpyAsync(s_out, s_in, (size_t)L_ * S_ * 4, hipMemcpyDeviceToDevice, stream);
  hipMemcpyAsync(z_out, z_in, (size_t)LL_ * P_ * 4, hipMemcpyDeviceToDevice, stream);

  // ---- PairToSequence bias (from input pairwise state) ----
  {
    h16* pn = (h16*)(ws + OFF_ZN);
    const h16* Wp2s = conv(P_P2S_LIN_W, 16, 128, 16);
    k_ln<<<LL_, 128, 0, stream>>>(z_in, PRM[P_P2S_LN_G], PRM[P_P2S_LN_B], pn, 128);
    gemm(stream, pn, 128, Wp2s, 16, nullptr, biasS, 16, LL_, 16, 128, 0, M_F32);
  }

  // ---- gated sequence attention ----
  {
    const h16* Wproj = conv(P_ATTN_PROJ_W, 3072, 1024, 3072);
    const h16* Wg    = conv(P_ATTN_G_W, 1024, 1024, 1024);
    const h16* Wo    = conv(P_ATTN_O_W, 1024, 1024, 1024);
    k_ln<<<L_, 128, 0, stream>>>(s_in, PRM[P_LN1_G], PRM[P_LN1_B], y16, 1024);
    gemm(stream, y16, 1024, Wproj, 3072, nullptr, qkv, 3072, L_, 3072, 1024, 0, M_F16);
    k_seq_kt<<<1024, 256, 0, stream>>>(qkv, kts);
    k_seq_flash<<<dim3(4, 16), 128, 0, stream>>>(qkv, kts, biasS, mask, oS);
    gemm(stream, y16, 1024, Wg, 1024, PRM[P_ATTN_G_B], gS, 1024, L_, 1024, 1024, 2, M_F16);
    k_mul16<<<1024, 256, 0, stream>>>(oS, gS, ogS, (size_t)L_ * 1024);
    gemm(stream, ogS, 1024, Wo, 1024, PRM[P_ATTN_O_B], s_out, 1024, L_, 1024, 1024, 0, M_F32ADD);
  }

  // ---- sequence residue MLP ----
  {
    const h16* W1 = conv(P_MS_W1_W, 4096, 1024, 4096);
    const h16* W2 = conv(P_MS_W2_W, 1024, 4096, 1024);
    k_ln<<<L_, 128, 0, stream>>>(s_out, PRM[P_MS_LN_G], PRM[P_MS_LN_B], x16s, 1024);
    gemm(stream, x16s, 1024, W1, 4096, PRM[P_MS_W1_B], h4k, 4096, L_, 4096, 1024, 1, M_F16);
    gemm(stream, h4k, 4096, W2, 1024, PRM[P_MS_W2_B], s_out, 1024, L_, 1024, 4096, 0, M_F32ADD);
  }

  // ---- SequenceToPair ----
  {
    const h16* Wp = conv(P_S2P_PROJ_W, 128, 1024, 128);
    const h16* Wo = conv(P_S2P_O_W, 128, 128, 128);
    h16* F = (h16*)(ws + OFF_T1);
    k_ln<<<L_, 128, 0, stream>>>(s_out, PRM[P_S2P_LN_G], PRM[P_S2P_LN_B], x16s, 1024);
    gemm(stream, x16s, 1024, Wp, 128, PRM[P_S2P_PROJ_B], sp, 128, L_, 128, 1024, 0, M_F32);
    k_s2p_feat<<<2048, 256, 0, stream>>>(sp, F);
    gemm(stream, F, 128, Wo, 128, PRM[P_S2P_O_B], z_out, 128, LL_, 128, 128, 0, M_F32ADD);
  }

  // ---- triangle multiplicative updates ----
  auto do_trimul = [&](int base, int aswap, int bswap) {
    const h16* Wag = conv(base + 1, 128, 128, 128);
    const h16* Wap = conv(base + 3, 128, 128, 128);
    const h16* Wbg = conv(base + 5, 128, 128, 128);
    const h16* Wbp = conv(base + 7, 128, 128, 128);
    const h16* Wg  = conv(base + 9, 128, 128, 128);
    const h16* Wz  = conv(base + 15, 128, 128, 128);
    h16* zn = (h16*)(ws + OFF_ZN); h16* t1 = (h16*)(ws + OFF_T1);
    h16* t2 = (h16*)(ws + OFF_T2); h16* acm = (h16*)(ws + OFF_ACM);
    h16* bcm = (h16*)(ws + OFF_BCM); float* xt = (float*)(ws + OFF_XT);
    h16* xn = (h16*)(ws + OFF_XN); float* zu = (float*)(ws + OFF_ZU);
    h16* g16 = (h16*)(ws + OFF_G16);
    k_ln<<<LL_, 128, 0, stream>>>(z_out, PRM[base + 11], PRM[base + 10], zn, 128);
    gemm(stream, zn, 128, Wag, 128, PRM[base + 0], t1, 128, LL_, 128, 128, 2, M_F16);
    gemm(stream, zn, 128, Wap, 128, PRM[base + 2], t2, 128, LL_, 128, 128, 0, M_F16);
    k_combine<<<2048, 256, 0, stream>>>(t1, t2, mask, acm, aswap);
    gemm(stream, zn, 128, Wbg, 128, PRM[base + 4], t1, 128, LL_, 128, 128, 2, M_F16);
    gemm(stream, zn, 128, Wbp, 128, PRM[base + 6], t2, 128, LL_, 128, 128, 0, M_F16);
    k_combine<<<2048, 256, 0, stream>>>(t1, t2, mask, bcm, bswap);
    k_trimul<<<dim3(16, 4, 128), 128, 0, stream>>>(acm, bcm, xt);
    k_ln<<<LL_, 128, 0, stream>>>(xt, PRM[base + 13], PRM[base + 12], xn, 128);
    gemm(stream, xn, 128, Wz, 128, PRM[base + 14], zu, 128, LL_, 128, 128, 0, M_F32);
    gemm(stream, zn, 128, Wg, 128, PRM[base + 8], g16, 128, LL_, 128, 128, 2, M_F16);
    k_fma32<<<2048, 256, 0, stream>>>(z_out, zu, g16, (size_t)LL_ * 128);
  };
  do_trimul(T_TMO, /*a: t=i*L+k*/ 0, /*b: store [c][k][j] from t=j*L+k*/ 1);
  do_trimul(T_TMI, /*a: t=k*L+i*/ 1, /*b: t=k*L+j*/ 0);

  // ---- triangle attention (starting / ending node) ----
  auto do_triatt = [&](int base, const float* zin, bool add_trans) {
    const h16* Wb = conv(base + 0, 4, 128, 16);
    const h16* Wg = conv(base + 2, 128, 128, 128);
    const h16* Wk = conv(base + 3, 128, 128, 128);
    const h16* Wo = conv(base + 7, 128, 128, 128);
    const h16* Wq = conv(base + 8, 128, 128, 128);
    const h16* Wv = conv(base + 9, 128, 128, 128);
    h16* zn = (h16*)(ws + OFF_ZN); h16* q16 = (h16*)(ws + OFF_T1);
    h16* v16 = (h16*)(ws + OFF_T2); h16* kT = (h16*)(ws + OFF_ACM);
    h16* o16 = (h16*)(ws + OFF_XN); h16* g16 = (h16*)(ws + OFF_G16);
    h16* og = (h16*)(ws + OFF_BCM);
    k_ln<<<LL_, 128, 0, stream>>>(zin, PRM[base + 5], PRM[base + 4], zn, 128);
    gemm(stream, zn, 128, Wq, 128, nullptr, q16, 128, LL_, 128, 128, 0, M_F16);
    gemm(stream, zn, 128, Wk, 128, nullptr, kT, LL_, LL_, 128, 128, 0, M_F16T);
    gemm(stream, zn, 128, Wv, 128, nullptr, v16, 128, LL_, 128, 128, 0, M_F16);
    gemm(stream, zn, 128, Wb, 16, nullptr, bias4, 4, LL_, 4, 128, 0, M_F32);
    k_tri_flash<<<dim3(4, 4, 256), 128, 0, stream>>>(q16, kT, v16, bias4, mask, o16);
    gemm(stream, zn, 128, Wg, 128, PRM[base + 1], g16, 128, LL_, 128, 128, 2, M_F16);
    k_mul16<<<2048, 256, 0, stream>>>(o16, g16, og, (size_t)LL_ * 128);
    if (!add_trans) {
      gemm(stream, og, 128, Wo, 128, PRM[base + 6], z_out, 128, LL_, 128, 128, 0, M_F32ADD);
    } else {
      float* upd = (float*)(ws + OFF_XT);
      gemm(stream, og, 128, Wo, 128, PRM[base + 6], upd, 128, LL_, 128, 128, 0, M_F32);
      k_swap_add<<<2048, 256, 0, stream>>>(z_out, upd);
    }
  };
  do_triatt(T_TAS, z_out, false);
  {
    float* zT = (float*)(ws + OFF_ZU);
    k_swap_copy<<<2048, 256, 0, stream>>>(z_out, zT);
    do_triatt(T_TAE, zT, true);
  }

  // ---- pair residue MLP (fp16) ----
  {
    const h16* W1 = conv(P_MP_W1_W, 512, 128, 512);
    const h16* W2 = conv(P_MP_W2_W, 128, 512, 128);
    h16* zn = (h16*)(ws + OFF_ZN);
    h16* hid = (h16*)(ws + OFF_ACM);  // [LL][512] f16 (slots 3-6, free here)
    k_ln<<<LL_, 128, 0, stream>>>(z_out, PRM[P_MP_LN_G], PRM[P_MP_LN_B], zn, 128);
    gemm(stream, zn, 128, W1, 512, PRM[P_MP_W1_B], hid, 512, LL_, 512, 128, 1, M_F16);
    gemm(stream, hid, 512, W2, 128, PRM[P_MP_W2_B], z_out, 128, LL_, 128, 512, 0, M_F32ADD);
  }
}
